// BidirectionalBPLayer_42709154792032
// MI455X (gfx1250) — matrix operand order repaired and checked by reading, not compiled, not run on hardware
//
#include <hip/hip_runtime.h>
#include <cstdint>
#include <cstddef>

// ---------------- problem constants (fixed by the reference) ----------------
#define HIDC   128
#define GOUTC  512          // HEADS * HID
#define NHEADS 4
static constexpr int NVAR = 65536;
static constexpr int NCHK = 32768;
static constexpr int NTOT = 98304;   // NVAR + NCHK
static constexpr int NE   = 131072;  // edges per direction
static constexpr int KPAD = 160;     // zero-padded K for the 129/144-wide GEMMs

typedef __attribute__((ext_vector_type(16))) _Float16 v16h;
typedef __attribute__((ext_vector_type(8)))  _Float16 v8h;
typedef __attribute__((ext_vector_type(8)))  float    v8f;

// ---------------- input index map ----------------
// Top level: setup_inputs() dict insertion order.
// 'params' flattened as a jax pytree: dict keys sorted alphabetically, recursively.
enum {
  IN_X = 0, IN_V2C_EI, IN_C2V_EI, IN_EDGE_ATTR, IN_DEG, IN_LLR, IN_VMASK, IN_CMASK, IN_NVAR,
  // params['c2v']
  P_C2V_GAT_ATT, P_C2V_GAT_LINE, P_C2V_GAT_LINL, P_C2V_GAT_LINR,
  P_C2V_GATE_EMBED, P_C2V_GATE_G1B, P_C2V_GATE_G1W, P_C2V_GATE_G2B, P_C2V_GATE_G2W,
  P_C2V_GATE_LNB, P_C2V_GATE_LNG,
  P_C2V_LNB, P_C2V_LNG, P_C2V_T1B, P_C2V_T1W, P_C2V_T2B, P_C2V_T2W, P_C2V_TLNB, P_C2V_TLNG,
  P_C2V_RES,
  P_CGRU_BHH, P_CGRU_BIH, P_CGRU_WHH, P_CGRU_WIH,
  // params['v2c']
  P_V2C_GAT_ATT, P_V2C_GAT_LINE, P_V2C_GAT_LINL, P_V2C_GAT_LINR,
  P_V2C_GATE_EMBED, P_V2C_GATE_G1B, P_V2C_GATE_G1W, P_V2C_GATE_G2B, P_V2C_GATE_G2W,
  P_V2C_GATE_LNB, P_V2C_GATE_LNG,
  P_V2C_LLRB, P_V2C_LLRLNB, P_V2C_LLRLNG, P_V2C_LLRW, P_V2C_LNB, P_V2C_LNG,
  P_V2C_RES,
  P_VGRU_BHH, P_VGRU_BIH, P_VGRU_WHH, P_VGRU_WIH
};

// ---------------- WMMA fragment loaders (branch-free; all K pre-padded) ----------------
// A-matrix 16x32 f16 layout (ISA 7.12.2): lanes 0-15 hold M=0..15 with K {0-7,16-23},
// lanes 16-31 hold M=0..15 with K {8-15,24-31}.  A is f32 in memory; cvt on the fly.
__device__ __forceinline__ v16h frag_load_a(const float* __restrict__ base, int row,
                                            int lda, int kb) {
  const float* p = base + (size_t)row * lda + kb + ((threadIdx.x & 16) ? 8 : 0);
  const float4 a0 = *(const float4*)(p);
  const float4 a1 = *(const float4*)(p + 4);
  const float4 b0 = *(const float4*)(p + 16);
  const float4 b1 = *(const float4*)(p + 20);
  v16h f;
  f[0]=(_Float16)a0.x;  f[1]=(_Float16)a0.y;  f[2]=(_Float16)a0.z;  f[3]=(_Float16)a0.w;
  f[4]=(_Float16)a1.x;  f[5]=(_Float16)a1.y;  f[6]=(_Float16)a1.z;  f[7]=(_Float16)a1.w;
  f[8]=(_Float16)b0.x;  f[9]=(_Float16)b0.y;  f[10]=(_Float16)b0.z; f[11]=(_Float16)b0.w;
  f[12]=(_Float16)b1.x; f[13]=(_Float16)b1.y; f[14]=(_Float16)b1.z; f[15]=(_Float16)b1.w;
  return f;
}

// B-matrix 32x16 f16 layout (matches A): lane holds column
// n = lane&15; lanes 0-15 hold K {0-7,16-23}, lanes 16-31 hold K {8-15,24-31}.
// Weights are pre-packed to f16 (Nout, K) row-major, so row n == column n of B = W^T.
__device__ __forceinline__ v16h frag_load_b(const _Float16* __restrict__ base, int row,
                                            int ldb, int kb) {
  const _Float16* p = base + (size_t)row * ldb + kb + ((threadIdx.x & 16) ? 8 : 0);
  const v8h lo = *(const v8h*)(p);
  const v8h hi = *(const v8h*)(p + 16);
  v16h f;
#pragma unroll
  for (int i = 0; i < 8; ++i) { f[i] = lo[i]; f[8 + i] = hi[i]; }
  return f;
}

// ---------------- generic WMMA GEMM: C = act(A[M,K] @ W[Nout,K]^T + bias) ----------------
// block = 128 threads = 4 waves; wave w owns output tile (blockIdx.x, blockIdx.y*4+w).
// K compile-time (multiple of 32), M % 16 == 0, Nout % 64 == 0 at every call site.
// ACT: 0=none, 1=relu, 2=sigmoid, 3=tanh
template <int K, int ACT>
__global__ __launch_bounds__(128)
void gemm_wmma(const float* __restrict__ A, int lda,
               const _Float16* __restrict__ Wh,         // (Nout, K) f16 row-major
               const float* __restrict__ bias,          // (Nout) or nullptr
               float* __restrict__ C, int ldc) {
  const int wave = threadIdx.x >> 5;
  const int lane = threadIdx.x & 31;
  const int row0 = blockIdx.x * 16;
  const int col0 = (blockIdx.y * 4 + wave) * 16;
  v8f acc = {};
#pragma unroll
  for (int kb = 0; kb < K; kb += 32) {
    v16h a = frag_load_a(A, row0 + (lane & 15), lda, kb);
    v16h b = frag_load_b(Wh, col0 + (lane & 15), K, kb);
    acc = __builtin_amdgcn_wmma_f32_16x16x32_f16(false, a, false, b, (short)0, acc,
                                                 false, false);
  }
  const int n  = col0 + (lane & 15);
  const int mb = row0 + ((threadIdx.x & 16) ? 8 : 0);
  const float bv = bias ? bias[n] : 0.0f;
#pragma unroll
  for (int r = 0; r < 8; ++r) {
    float v = acc[r] + bv;
    if (ACT == 1)      v = fmaxf(v, 0.0f);
    else if (ACT == 2) v = 1.0f / (1.0f + __expf(-v));
    else if (ACT == 3) v = tanhf(v);
    C[(size_t)(mb + r) * ldc + n] = v;
  }
}

// ---------------- weight pack: f32 (Nout,K) -> f16 (Nout,Kpad), zero-padded ----------------
__global__ __launch_bounds__(256)
void pack_w_f16_kernel(const float* __restrict__ w, _Float16* __restrict__ out,
                       int Nout, int K, int Kpad) {
  const long long idx = (long long)blockIdx.x * 256 + threadIdx.x;
  if (idx >= (long long)Nout * Kpad) return;
  const int n = (int)(idx / Kpad), j = (int)(idx % Kpad);
  out[idx] = (j < K) ? (_Float16)w[(size_t)n * K + j] : (_Float16)0.0f;
}

// ---------------- LayerNorm over HID=128, one wave per row ----------------
// Y = (optional relu)( LN( X .* mul ) * g + b )
__global__ __launch_bounds__(256)
void ln_kernel(const float* __restrict__ X, const float* __restrict__ mul,
               const float* __restrict__ g, const float* __restrict__ b,
               float* __restrict__ Y, int M, int do_relu) {
  const int row = blockIdx.x * 8 + (threadIdx.x >> 5);
  if (row >= M) return;
  const int lane = threadIdx.x & 31;
  float4 v = *(const float4*)(X + (size_t)row * HIDC + lane * 4);
  if (mul) {
    float4 m4 = *(const float4*)(mul + (size_t)row * HIDC + lane * 4);
    v.x *= m4.x; v.y *= m4.y; v.z *= m4.z; v.w *= m4.w;
  }
  float s  = v.x + v.y + v.z + v.w;
  float ss = v.x*v.x + v.y*v.y + v.z*v.z + v.w*v.w;
#pragma unroll
  for (int off = 16; off > 0; off >>= 1) {
    s  += __shfl_xor(s, off, 32);
    ss += __shfl_xor(ss, off, 32);
  }
  const float mean = s * (1.0f / HIDC);
  const float var  = ss * (1.0f / HIDC) - mean * mean;
  const float inv  = rsqrtf(var + 1e-5f);
  float4 gg = *(const float4*)(g + lane * 4);
  float4 bb = *(const float4*)(b + lane * 4);
  float4 o;
  o.x = (v.x - mean) * inv * gg.x + bb.x;
  o.y = (v.y - mean) * inv * gg.y + bb.y;
  o.z = (v.z - mean) * inv * gg.z + bb.z;
  o.w = (v.w - mean) * inv * gg.w + bb.w;
  if (do_relu) {
    o.x = fmaxf(o.x, 0.f); o.y = fmaxf(o.y, 0.f);
    o.z = fmaxf(o.z, 0.f); o.w = fmaxf(o.w, 0.f);
  }
  *(float4*)(Y + (size_t)row * HIDC + lane * 4) = o;
}

// ---------------- monotonic float<->uint encoding for atomicMax segment-max ----------------
__device__ __forceinline__ unsigned fenc(float f) {
  unsigned u = __float_as_uint(f);
  return (u & 0x80000000u) ? ~u : (u | 0x80000000u);
}
__device__ __forceinline__ float fdec(unsigned u) {
  return (u & 0x80000000u) ? __uint_as_float(u & 0x7FFFFFFFu) : __uint_as_float(~u);
}
#define ENC_NEGINF 0x007FFFFFu   // fenc(-inf)

// ---------------- GATv2 edge pass 1: logits + segment max (wave per edge-head) ---------------
__global__ __launch_bounds__(256)
void edge_logits_kernel(const float* __restrict__ xl, const float* __restrict__ xr,
                        const float* __restrict__ ea,       // (E,8)
                        const float* __restrict__ lin_edge, // (512,8)
                        const float* __restrict__ att,      // (4,128)
                        const int* __restrict__ src, const int* __restrict__ dst,
                        float* __restrict__ logits, unsigned* __restrict__ amax) {
  const int task = blockIdx.x * 8 + (threadIdx.x >> 5);
  if (task >= NE * NHEADS) return;
  const int e = task >> 2, h = task & 3;
  const int lane = threadIdx.x & 31;
  const int s = src[e], d = dst[e];
  const float4 ea0 = *(const float4*)(ea + (size_t)e * 8);
  const float4 ea1 = *(const float4*)(ea + (size_t)e * 8 + 4);
  const int hc0 = h * HIDC + lane * 4;
  const float4 xlv = *(const float4*)(xl + (size_t)s * GOUTC + hc0);
  const float4 xrv = *(const float4*)(xr + (size_t)d * GOUTC + hc0);
  const float4 atv = *(const float4*)(att + hc0);
  float acc = 0.0f;
#pragma unroll
  for (int q = 0; q < 4; ++q) {
    const float4 le0 = *(const float4*)(lin_edge + (size_t)(hc0 + q) * 8);
    const float4 le1 = *(const float4*)(lin_edge + (size_t)(hc0 + q) * 8 + 4);
    float ep = ea0.x*le0.x + ea0.y*le0.y + ea0.z*le0.z + ea0.w*le0.w
             + ea1.x*le1.x + ea1.y*le1.y + ea1.z*le1.z + ea1.w*le1.w;
    const float xlq = (q == 0) ? xlv.x : (q == 1) ? xlv.y : (q == 2) ? xlv.z : xlv.w;
    const float xrq = (q == 0) ? xrv.x : (q == 1) ? xrv.y : (q == 2) ? xrv.z : xrv.w;
    const float atq = (q == 0) ? atv.x : (q == 1) ? atv.y : (q == 2) ? atv.z : atv.w;
    float m = xlq + xrq + ep;
    m = (m > 0.0f) ? m : 0.2f * m;   // leaky_relu(0.2)
    acc += m * atq;
  }
#pragma unroll
  for (int off = 16; off > 0; off >>= 1) acc += __shfl_xor(acc, off, 32);
  if (lane == 0) {
    logits[task] = acc;
    atomicMax(amax + (size_t)d * NHEADS + h, fenc(acc));
  }
}

// ---------------- GATv2 edge pass 2: softmax numerator scatter (wave per edge-head) ----------
__global__ __launch_bounds__(256)
void edge_accum_kernel(const float* __restrict__ xl, const float* __restrict__ logits,
                       const unsigned* __restrict__ amax,
                       const int* __restrict__ src, const int* __restrict__ dst,
                       float* __restrict__ denom, float* __restrict__ accum) {
  const int task = blockIdx.x * 8 + (threadIdx.x >> 5);
  if (task >= NE * NHEADS) return;
  const int e = task >> 2, h = task & 3;
  const int lane = threadIdx.x & 31;
  const int s = src[e], d = dst[e];
  const float am = fdec(amax[(size_t)d * NHEADS + h]);
  const float ex = __expf(logits[task] - am);
  if (lane == 0) atomicAdd(denom + (size_t)d * NHEADS + h, ex);
  const int hc0 = h * HIDC + lane * 4;
  const float4 xlv = *(const float4*)(xl + (size_t)s * GOUTC + hc0);
  float* dp = accum + (size_t)d * GOUTC + hc0;
  atomicAdd(dp + 0, ex * xlv.x);
  atomicAdd(dp + 1, ex * xlv.y);
  atomicAdd(dp + 2, ex * xlv.z);
  atomicAdd(dp + 3, ex * xlv.w);
}

// ---------------- head mean: out[n,c] = (1/4) sum_h accum[n,h,c]/max(denom,1e-16) ------------
__global__ __launch_bounds__(256)
void head_mean_kernel(const float* __restrict__ accum, const float* __restrict__ denom,
                      float* __restrict__ out) {
  const long long idx = (long long)blockIdx.x * 256 + threadIdx.x;
  if (idx >= (long long)NTOT * HIDC) return;
  const int n = (int)(idx >> 7), c = (int)(idx & 127);
  float s = 0.0f;
#pragma unroll
  for (int h = 0; h < NHEADS; ++h)
    s += accum[(size_t)n * GOUTC + h * HIDC + c] /
         fmaxf(denom[(size_t)n * NHEADS + h], 1e-16f);
  out[idx] = 0.25f * s;
}

// ---------------- pack kernels (activation side, zero-padded to KPAD) ----------------
__global__ __launch_bounds__(256)
void pack_llr_kernel(const float* __restrict__ x, const float* __restrict__ llr,
                     float* __restrict__ out) {            // (NVAR, KPAD)
  const long long idx = (long long)blockIdx.x * 256 + threadIdx.x;
  if (idx >= (long long)NVAR * KPAD) return;
  const int n = (int)(idx / KPAD), j = (int)(idx % KPAD);
  float v = 0.0f;
  if (j < HIDC)        v = x[(size_t)n * HIDC + j];
  else if (j == HIDC)  v = llr[n];
  out[idx] = v;
}

__global__ __launch_bounds__(256)
void pack_gatein_kernel(const float* __restrict__ h, const int* __restrict__ deg,
                        const float* __restrict__ embed,   // (100,16)
                        float* __restrict__ out) {          // (NTOT, KPAD)
  const long long idx = (long long)blockIdx.x * 256 + threadIdx.x;
  if (idx >= (long long)NTOT * KPAD) return;
  const int n = (int)(idx / KPAD), j = (int)(idx % KPAD);
  float v = 0.0f;
  if (j < HIDC) {
    v = h[(size_t)n * HIDC + j];
  } else if (j < 144) {
    int dg = deg[n]; dg = dg < 0 ? 0 : (dg > 99 ? 99 : dg);
    v = embed[(size_t)dg * 16 + (j - HIDC)];
  }
  out[idx] = v;
}

// ---------------- GRU combine: out = (1-z)*n + z*h ----------------
__global__ __launch_bounds__(256)
void gru_combine_kernel(const float* __restrict__ gi, const float* __restrict__ gh,
                        const float* __restrict__ hprev, float* __restrict__ out, int M) {
  const long long idx = (long long)blockIdx.x * 256 + threadIdx.x;
  if (idx >= (long long)M * HIDC) return;
  const int r = (int)(idx >> 7), c = (int)(idx & 127);
  const size_t b = (size_t)r * 384;
  const float rr = 1.0f / (1.0f + __expf(-(gi[b + c] + gh[b + c])));
  const float zz = 1.0f / (1.0f + __expf(-(gi[b + 128 + c] + gh[b + 128 + c])));
  const float nn = tanhf(gi[b + 256 + c] + rr * gh[b + 256 + c]);
  const float h = hprev[idx];
  out[idx] = (1.0f - zz) * nn + zz * h;
}

// ---------------- misc elementwise ----------------
__global__ __launch_bounds__(256)
void axpy_res_kernel(const float* __restrict__ x, const float* __restrict__ x0,
                     const float* __restrict__ res, float* __restrict__ out, long long n) {
  const long long idx = (long long)blockIdx.x * 256 + threadIdx.x;
  if (idx >= n) return;
  out[idx] = x[idx] + (*res) * x0[idx];
}

__global__ __launch_bounds__(256)
void fill_u32_kernel(unsigned* __restrict__ p, unsigned v, long long n) {
  const long long idx = (long long)blockIdx.x * 256 + threadIdx.x;
  if (idx < n) p[idx] = v;
}

// ---------------- host-side helpers ----------------
static inline int cdiv(long long a, long long b) { return (int)((a + b - 1) / b); }

struct GatParams {           // f16 packed weights + raw f32 params
  const _Float16 *lin_l, *lin_r;
  const float *lin_edge, *att;
  const float *g_embed; const _Float16 *g1w; const float *g1b;
  const float *gln_g, *gln_b; const _Float16 *g2w; const float *g2b;
  const float *ln_g, *ln_b;
};

struct Ws {
  float* xbuf;  float* xwork; float* xl; float* xr;      // xr doubles as accum
  float* hout;  float* gatein; float* gbuf1; float* gbuf2;
  float* logits; unsigned* amax; float* denom;
  float* gi; float* gh; float* catllr; float* tb1; float* tb2;
};

// One GATv2 direction + degree gate + LN. Output written to w.gbuf2.
static void run_gat(const float* xsrc, const GatParams& P,
                    const int* ei, const float* ea, const int* deg,
                    Ws& w, hipStream_t stream) {
  const int* src = ei;
  const int* dst = ei + NE;
  dim3 blk128(128), blk256(256);
  // xl = x @ lin_l^T ; xr = x @ lin_r^T
  gemm_wmma<HIDC, 0><<<dim3(NTOT / 16, GOUTC / 64), blk128, 0, stream>>>(
      xsrc, HIDC, P.lin_l, nullptr, w.xl, GOUTC);
  gemm_wmma<HIDC, 0><<<dim3(NTOT / 16, GOUTC / 64), blk128, 0, stream>>>(
      xsrc, HIDC, P.lin_r, nullptr, w.xr, GOUTC);
  // segment max init + logits
  fill_u32_kernel<<<cdiv((long long)NTOT * NHEADS, 256), blk256, 0, stream>>>(
      w.amax, ENC_NEGINF, (long long)NTOT * NHEADS);
  edge_logits_kernel<<<NE * NHEADS / 8, blk256, 0, stream>>>(
      w.xl, w.xr, ea, P.lin_edge, P.att, src, dst, w.logits, w.amax);
  // zero denom + accum (accum aliases xr — dead after logits pass)
  fill_u32_kernel<<<cdiv((long long)NTOT * NHEADS, 256), blk256, 0, stream>>>(
      (unsigned*)w.denom, 0u, (long long)NTOT * NHEADS);
  fill_u32_kernel<<<cdiv((long long)NTOT * GOUTC, 256), blk256, 0, stream>>>(
      (unsigned*)w.xr, 0u, (long long)NTOT * GOUTC);
  edge_accum_kernel<<<NE * NHEADS / 8, blk256, 0, stream>>>(
      w.xl, w.logits, w.amax, src, dst, w.denom, w.xr);
  head_mean_kernel<<<cdiv((long long)NTOT * HIDC, 256), blk256, 0, stream>>>(
      w.xr, w.denom, w.hout);
  // degree gate
  pack_gatein_kernel<<<cdiv((long long)NTOT * KPAD, 256), blk256, 0, stream>>>(
      w.hout, deg, P.g_embed, w.gatein);
  gemm_wmma<KPAD, 0><<<dim3(NTOT / 16, HIDC / 64), blk128, 0, stream>>>(
      w.gatein, KPAD, P.g1w, P.g1b, w.gbuf1, HIDC);
  ln_kernel<<<cdiv(NTOT, 8), blk256, 0, stream>>>(
      w.gbuf1, nullptr, P.gln_g, P.gln_b, w.gbuf2, NTOT, 1);
  gemm_wmma<HIDC, 2><<<dim3(NTOT / 16, HIDC / 64), blk128, 0, stream>>>(
      w.gbuf2, HIDC, P.g2w, P.g2b, w.gbuf1, HIDC);   // sigmoid gate
  // x_dir = LN(h * gate) -> gbuf2
  ln_kernel<<<cdiv(NTOT, 8), blk256, 0, stream>>>(
      w.hout, w.gbuf1, P.ln_g, P.ln_b, w.gbuf2, NTOT, 0);
}

extern "C" void kernel_launch(void* const* d_in, const int* in_sizes, int n_in,
                              void* d_out, int out_size, void* d_ws, size_t ws_size,
                              hipStream_t stream) {
  (void)in_sizes; (void)n_in; (void)out_size; (void)ws_size;
  auto F = [&](int i) { return (const float*)d_in[i]; };
  auto I = [&](int i) { return (const int*)d_in[i]; };

  // bump-allocate workspace (256B aligned)
  char* p = (char*)d_ws;
  auto grab = [&](size_t nfloats) -> float* {
    p = (char*)(((uintptr_t)p + 255) & ~(uintptr_t)255);
    float* r = (float*)p;
    p += nfloats * sizeof(float);
    return r;
  };
  auto grabh = [&](size_t nhalf) -> _Float16* {
    p = (char*)(((uintptr_t)p + 255) & ~(uintptr_t)255);
    _Float16* r = (_Float16*)p;
    p += nhalf * sizeof(_Float16);
    return r;
  };
  Ws w;
  w.xbuf   = grab((size_t)NTOT * HIDC);
  w.xwork  = grab((size_t)NTOT * HIDC);
  w.xl     = grab((size_t)NTOT * GOUTC);
  w.xr     = grab((size_t)NTOT * GOUTC);
  w.hout   = grab((size_t)NTOT * HIDC);
  w.gatein = grab((size_t)NTOT * KPAD);
  w.gbuf1  = grab((size_t)NTOT * HIDC);
  w.gbuf2  = grab((size_t)NTOT * HIDC);
  w.logits = grab((size_t)NE * NHEADS);
  w.amax   = (unsigned*)grab((size_t)NTOT * NHEADS);
  w.denom  = grab((size_t)NTOT * NHEADS);
  w.gi     = grab((size_t)NVAR * 384);
  w.gh     = grab((size_t)NVAR * 384);
  w.catllr = grab((size_t)NVAR * KPAD);
  w.tb1    = grab((size_t)NCHK * HIDC);
  w.tb2    = grab((size_t)NCHK * HIDC);

  // f16 packed weights
  _Float16* v_linl = grabh((size_t)GOUTC * HIDC);
  _Float16* v_linr = grabh((size_t)GOUTC * HIDC);
  _Float16* c_linl = grabh((size_t)GOUTC * HIDC);
  _Float16* c_linr = grabh((size_t)GOUTC * HIDC);
  _Float16* v_g1w  = grabh((size_t)HIDC * KPAD);
  _Float16* c_g1w  = grabh((size_t)HIDC * KPAD);
  _Float16* v_g2w  = grabh((size_t)HIDC * HIDC);
  _Float16* c_g2w  = grabh((size_t)HIDC * HIDC);
  _Float16* llrw   = grabh((size_t)HIDC * KPAD);
  _Float16* t1w    = grabh((size_t)HIDC * HIDC);
  _Float16* t2w    = grabh((size_t)HIDC * HIDC);
  _Float16* cg_wih = grabh((size_t)384 * HIDC);
  _Float16* cg_whh = grabh((size_t)384 * HIDC);
  _Float16* vg_wih = grabh((size_t)384 * HIDC);
  _Float16* vg_whh = grabh((size_t)384 * HIDC);

  dim3 blk128(128), blk256(256);
  const float* x0 = F(IN_X);
  float* out = (float*)d_out;

  auto packw = [&](const float* src, _Float16* dst, int Nout, int K, int Kpad) {
    pack_w_f16_kernel<<<cdiv((long long)Nout * Kpad, 256), blk256, 0, stream>>>(
        src, dst, Nout, K, Kpad);
  };
  packw(F(P_V2C_GAT_LINL), v_linl, GOUTC, HIDC, HIDC);
  packw(F(P_V2C_GAT_LINR), v_linr, GOUTC, HIDC, HIDC);
  packw(F(P_C2V_GAT_LINL), c_linl, GOUTC, HIDC, HIDC);
  packw(F(P_C2V_GAT_LINR), c_linr, GOUTC, HIDC, HIDC);
  packw(F(P_V2C_GATE_G1W), v_g1w, HIDC, 144, KPAD);
  packw(F(P_C2V_GATE_G1W), c_g1w, HIDC, 144, KPAD);
  packw(F(P_V2C_GATE_G2W), v_g2w, HIDC, HIDC, HIDC);
  packw(F(P_C2V_GATE_G2W), c_g2w, HIDC, HIDC, HIDC);
  packw(F(P_V2C_LLRW),     llrw,  HIDC, 129, KPAD);
  packw(F(P_C2V_T1W),      t1w,   HIDC, HIDC, HIDC);
  packw(F(P_C2V_T2W),      t2w,   HIDC, HIDC, HIDC);
  packw(F(P_CGRU_WIH),     cg_wih, 384, HIDC, HIDC);
  packw(F(P_CGRU_WHH),     cg_whh, 384, HIDC, HIDC);
  packw(F(P_VGRU_WIH),     vg_wih, 384, HIDC, HIDC);
  packw(F(P_VGRU_WHH),     vg_whh, 384, HIDC, HIDC);

  // x (mutable working copy)
  hipMemcpyAsync(w.xbuf, x0, (size_t)NTOT * HIDC * sizeof(float),
                 hipMemcpyDeviceToDevice, stream);

  // ---------------- V2C ----------------
  // LLR fusion on variable nodes
  pack_llr_kernel<<<cdiv((long long)NVAR * KPAD, 256), blk256, 0, stream>>>(
      x0, F(IN_LLR), w.catllr);
  gemm_wmma<KPAD, 0><<<dim3(NVAR / 16, HIDC / 64), blk128, 0, stream>>>(
      w.catllr, KPAD, llrw, F(P_V2C_LLRB), w.gbuf1, HIDC);
  ln_kernel<<<cdiv(NVAR, 8), blk256, 0, stream>>>(
      w.gbuf1, nullptr, F(P_V2C_LLRLNG), F(P_V2C_LLRLNB), w.xwork, NVAR, 1);
  hipMemcpyAsync(w.xwork + (size_t)NVAR * HIDC, w.xbuf + (size_t)NVAR * HIDC,
                 (size_t)NCHK * HIDC * sizeof(float), hipMemcpyDeviceToDevice, stream);

  GatParams pv = { v_linl, v_linr, F(P_V2C_GAT_LINE), F(P_V2C_GAT_ATT),
                   F(P_V2C_GATE_EMBED), v_g1w, F(P_V2C_GATE_G1B),
                   F(P_V2C_GATE_LNG), F(P_V2C_GATE_LNB), v_g2w, F(P_V2C_GATE_G2B),
                   F(P_V2C_LNG), F(P_V2C_LNB) };
  run_gat(w.xwork, pv, I(IN_V2C_EI), F(IN_EDGE_ATTR), I(IN_DEG), w, stream);
  // check-node GRU: xin = x_v2c[NVAR:], h = x[NVAR:]
  gemm_wmma<HIDC, 0><<<dim3(NCHK / 16, 384 / 64), blk128, 0, stream>>>(
      w.gbuf2 + (size_t)NVAR * HIDC, HIDC, cg_wih, F(P_CGRU_BIH), w.gi, 384);
  gemm_wmma<HIDC, 0><<<dim3(NCHK / 16, 384 / 64), blk128, 0, stream>>>(
      w.xbuf + (size_t)NVAR * HIDC, HIDC, cg_whh, F(P_CGRU_BHH), w.gh, 384);
  gru_combine_kernel<<<cdiv((long long)NCHK * HIDC, 256), blk256, 0, stream>>>(
      w.gi, w.gh, w.xbuf + (size_t)NVAR * HIDC, w.xbuf + (size_t)NVAR * HIDC, NCHK);
  // x = concat(x[:NVAR], new_check) + v2c_residual * x0  (in place)
  axpy_res_kernel<<<cdiv((long long)NTOT * HIDC, 256), blk256, 0, stream>>>(
      w.xbuf, x0, F(P_V2C_RES), w.xbuf, (long long)NTOT * HIDC);

  // ---------------- C2V ----------------
  // t = LN( tanh(xc @ t1^T + b1) @ t2^T + b2 )
  gemm_wmma<HIDC, 3><<<dim3(NCHK / 16, HIDC / 64), blk128, 0, stream>>>(
      w.xbuf + (size_t)NVAR * HIDC, HIDC, t1w, F(P_C2V_T1B), w.tb1, HIDC);
  gemm_wmma<HIDC, 0><<<dim3(NCHK / 16, HIDC / 64), blk128, 0, stream>>>(
      w.tb1, HIDC, t2w, F(P_C2V_T2B), w.tb2, HIDC);
  ln_kernel<<<cdiv(NCHK, 8), blk256, 0, stream>>>(
      w.tb2, nullptr, F(P_C2V_TLNG), F(P_C2V_TLNB),
      w.xwork + (size_t)NVAR * HIDC, NCHK, 0);
  hipMemcpyAsync(w.xwork, w.xbuf, (size_t)NVAR * HIDC * sizeof(float),
                 hipMemcpyDeviceToDevice, stream);

  GatParams pc = { c_linl, c_linr, F(P_C2V_GAT_LINE), F(P_C2V_GAT_ATT),
                   F(P_C2V_GATE_EMBED), c_g1w, F(P_C2V_GATE_G1B),
                   F(P_C2V_GATE_LNG), F(P_C2V_GATE_LNB), c_g2w, F(P_C2V_GATE_G2B),
                   F(P_C2V_LNG), F(P_C2V_LNB) };
  run_gat(w.xwork, pc, I(IN_C2V_EI), F(IN_EDGE_ATTR) + (size_t)NE * 8, I(IN_DEG), w, stream);
  // variable-node GRU: xin = x_c2v[:NVAR], h = x[:NVAR]
  gemm_wmma<HIDC, 0><<<dim3(NVAR / 16, 384 / 64), blk128, 0, stream>>>(
      w.gbuf2, HIDC, vg_wih, F(P_VGRU_BIH), w.gi, 384);
  gemm_wmma<HIDC, 0><<<dim3(NVAR / 16, 384 / 64), blk128, 0, stream>>>(
      w.xbuf, HIDC, vg_whh, F(P_VGRU_BHH), w.gh, 384);
  gru_combine_kernel<<<cdiv((long long)NVAR * HIDC, 256), blk256, 0, stream>>>(
      w.gi, w.gh, w.xbuf, w.xbuf, NVAR);
  // out = x + c2v_residual * x0
  axpy_res_kernel<<<cdiv((long long)NTOT * HIDC, 256), blk256, 0, stream>>>(
      w.xbuf, x0, F(P_C2V_RES), out, (long long)NTOT * HIDC);
}
